// Attention_21586505629971
// MI455X (gfx1250) — compile-verified, matrix operand order checked by reading, not run on hardware
//
#include <hip/hip_runtime.h>
#include <hip/hip_bf16.h>
#include <math.h>

typedef __bf16 bf16_t;
typedef __attribute__((ext_vector_type(16))) __bf16 v16bf;
typedef __attribute__((ext_vector_type(8)))  __bf16 v8bf;
typedef __attribute__((ext_vector_type(8)))  float  v8f;

typedef unsigned int uint32x4 __attribute__((ext_vector_type(4)));
typedef int          int32x8  __attribute__((ext_vector_type(8)));
typedef int          int32x4  __attribute__((ext_vector_type(4)));

#define B_  2
#define S_  2048
#define D_  2048
#define H_  16
#define HD_ 128
#define M_  (B_ * S_)

#define NEG_BIG (-3.0e38f)

// ---------------------------------------------------------------- WMMA helpers

__device__ __forceinline__ v8f wmma_bf16(v16bf a, v16bf b, v8f c) {
  return __builtin_amdgcn_wmma_f32_16x16x32_bf16(false, a, false, b, (short)0, c,
                                                 false, false);
}

// 16x32 16-bit fragment (A- or B-style; identical per-lane layout on CDNA5:
// lane holds row (lane&15), K chunks {k0..k0+7, k0+16..k0+23}, k0=8*(lane>>4)).
__device__ __forceinline__ v16bf frag16x32(const bf16_t* tile, int stride, int lane) {
  const int r  = lane & 15;
  const int k0 = (lane >> 4) << 3;
  const bf16_t* p = tile + r * stride + k0;
  v8bf lo = *(const v8bf*)(p);
  v8bf hi = *(const v8bf*)(p + 16);
  v16bf o;
#pragma unroll
  for (int i = 0; i < 8; ++i) { o[i] = lo[i]; o[i + 8] = hi[i]; }
  return o;
}

// ---------------------------------------------------------------- TDM descriptor
// 2D tensor tile -> LDS with row padding (CDNA5 ISA ch.8, D# groups 0/1).
// data_size = 2 bytes (bf16). Groups 2/3 zero (tile_dim2/3 = 0 -> 2D).
//  - lds_addr: LDS byte offset (low 32 bits of a flat __shared__ pointer)
//  - pad_int_code: pad every 2<<code DWORDs; pad_amt_code: insert code+1 DWORDs
__device__ __forceinline__ void tdm_load_2d(unsigned lds_byte_addr, const void* gptr,
                                            unsigned tile_d0, unsigned tile_d1,
                                            unsigned tensor_d0, unsigned tensor_d1,
                                            unsigned stride0,
                                            unsigned pad_int_code, unsigned pad_amt_code) {
  const unsigned long long ga = (unsigned long long)(size_t)gptr;   // 57-bit byte addr
  union { unsigned u[4]; uint32x4 v; } g0;
  g0.u[0] = 1u;                                             // count=1, user D#
  g0.u[1] = lds_byte_addr;                                  // LDS dest (bytes)
  g0.u[2] = (unsigned)ga;                                   // global_addr[31:0]
  g0.u[3] = (unsigned)((ga >> 32) & 0x1ffffffu) | (2u << 30); // addr[56:32] | type=2
  union { unsigned long long q[4]; int32x8 v; } g1;
  const unsigned long long w0 = (1ull << 16)                // data_size = 2 bytes
                              | (1ull << 20)                // pad_enable
                              | ((unsigned long long)pad_int_code << 22)
                              | ((unsigned long long)pad_amt_code << 25);
  g1.q[0] = w0 | ((unsigned long long)(tensor_d0 & 0xffffu) << 48);
  g1.q[1] = (unsigned long long)(tensor_d0 >> 16)
          | ((unsigned long long)tensor_d1 << 16)
          | ((unsigned long long)(tile_d0 & 0xffffu) << 48);
  g1.q[2] = (unsigned long long)tile_d1                     // tile_dim1; tile_dim2=0
          | ((unsigned long long)stride0 << 32);            // dim0_stride[31:0]
  g1.q[3] = ((unsigned long long)stride0 >> 32);            // dim0_stride[47:32]; dim1_stride=0
  const int32x4 gz4 = {0, 0, 0, 0};
  const int32x8 gz8 = {0, 0, 0, 0, 0, 0, 0, 0};
  __builtin_amdgcn_tensor_load_to_lds(g0.v, g1.v, gz4, gz4, gz8, 0);
}

__device__ __forceinline__ unsigned lds_off(const void* p) {
  return (unsigned)(size_t)p;   // flat LDS address: low 32 bits = LDS byte offset
}

// ---------------------------------------------------------------- convert f32 -> bf16

__global__ __launch_bounds__(256) void f32_to_bf16_kernel(const float* __restrict__ in,
                                                          bf16_t* __restrict__ out,
                                                          int n) {
  int i = (blockIdx.x * 256 + threadIdx.x) * 4;
  if (i < n) {
    float4 v = *(const float4*)(in + i);
    out[i + 0] = (bf16_t)v.x;
    out[i + 1] = (bf16_t)v.y;
    out[i + 2] = (bf16_t)v.z;
    out[i + 3] = (bf16_t)v.w;
  }
}

// ---------------------------------------------------------------- rotary f32 -> bf16

__global__ __launch_bounds__(256) void rotary_bf16_kernel(const float* __restrict__ X,
                                                          const float* __restrict__ C,
                                                          const float* __restrict__ Sn,
                                                          bf16_t* __restrict__ out) {
  const int i = blockIdx.x * 256 + threadIdx.x;   // pair index
  const int p = i & 63;
  const int s = (i >> 10) & (S_ - 1);
  const size_t off = (size_t)i * 2;
  const float x1 = X[off], x2 = X[off + 1];
  const float c = C[s * 64 + p], sn = Sn[s * 64 + p];
  out[off]     = (bf16_t)(x1 * c - x2 * sn);
  out[off + 1] = (bf16_t)(x1 * sn + x2 * c);
}

// ---------------------------------------------------------------- bf16 GEMM
// C[m,n] = sum_k A[m,k] * W[n,k].  Block tile 128x256, 8 waves, wave tile 64x64
// (4x4 WMMA).  K tiles DMA'd by the Tensor Data Mover into double-buffered LDS
// (TDM pads each 16-DWORD row with 4 DWORDs -> conflict-free stride 40).

#define GSTR  40   // 32 + 8 pad (matches TDM pad_int=3 (16 DW), pad_amt=3 (4 DW))

__global__ __launch_bounds__(256) void gemm_bf16_kernel(const bf16_t* __restrict__ A,
                                                        const bf16_t* __restrict__ W,
                                                        float* __restrict__ C) {
  __shared__ __align__(16) bf16_t As[2][128 * GSTR];   // 2 x 10 KB
  __shared__ __align__(16) bf16_t Bs[2][256 * GSTR];   // 2 x 20 KB

  const int tid  = threadIdx.x;
  const int lane = tid & 31;
  const int wave = tid >> 5;
  const int bm = blockIdx.y * 128;
  const int bn = blockIdx.x * 256;
  const int wm = (wave >> 2) * 64;   // 0 or 64
  const int wn = (wave & 3) * 64;    // 0,64,128,192

  // ---- TDM: post k-tile 0 into buffer 0 ----
  if (wave == 0) {
    tdm_load_2d(lds_off(&As[0][0]), A + (size_t)bm * D_, 32, 128, D_, M_, D_, 3, 3);
    tdm_load_2d(lds_off(&Bs[0][0]), W + (size_t)bn * D_, 32, 256, D_, D_, D_, 3, 3);
    __builtin_amdgcn_s_wait_tensorcnt(0);
  }
  __syncthreads();

  v8f acc[4][4] = {};
  int cur = 0;

  for (int k0 = 0; k0 < D_; k0 += 32) {
    const bool more = (k0 + 32) < D_;
    if (more && wave == 0) {       // post next k-tile while we compute this one
      tdm_load_2d(lds_off(&As[cur ^ 1][0]), A + (size_t)bm * D_ + k0 + 32,
                  32, 128, D_, M_, D_, 3, 3);
      tdm_load_2d(lds_off(&Bs[cur ^ 1][0]), W + (size_t)bn * D_ + k0 + 32,
                  32, 256, D_, D_, D_, 3, 3);
    }

    v16bf af[4], bfg[4];
#pragma unroll
    for (int i = 0; i < 4; ++i) af[i]  = frag16x32(&As[cur][(wm + i * 16) * GSTR], GSTR, lane);
#pragma unroll
    for (int j = 0; j < 4; ++j) bfg[j] = frag16x32(&Bs[cur][(wn + j * 16) * GSTR], GSTR, lane);
#pragma unroll
    for (int i = 0; i < 4; ++i)
#pragma unroll
      for (int j = 0; j < 4; ++j)
        acc[i][j] = wmma_bf16(af[i], bfg[j], acc[i][j]);

    if (more) {
      if (wave == 0) __builtin_amdgcn_s_wait_tensorcnt(0);  // next tile landed
      __syncthreads();
      cur ^= 1;
    }
  }

  // C epilogue: one base per wave, constant immediate offsets per store.
  const int half = lane >> 4, col = lane & 15;
  float* cb = C + (size_t)(bm + wm + 8 * half) * D_ + (bn + wn + col);
#pragma unroll
  for (int i = 0; i < 4; ++i)
#pragma unroll
    for (int j = 0; j < 4; ++j)
#pragma unroll
      for (int e = 0; e < 8; ++e)
        cb[(i * 16 + e) * D_ + j * 16] = acc[i][j][e];
}

// ---------------------------------------------------------------- flash attention
// Q,K,V: (B,S,H,hd) bf16.  O: (B,S,H,hd) fp32.  Causal, scale 1/sqrt(hd).
// Block = 8 waves = 128 q rows of one (b,h); wave owns 16 q rows.
// K tile via TDM (pad 4 DW per 64-DW row -> stride 136); V staged transposed.

#define KSTRK 136   // 128 + 8 pad (TDM pad_int=5 (64 DW), pad_amt=3 (4 DW))
#define VSTR  40    // 32 + 8 pad (V^T rows)
#define PSTR  40

__global__ __launch_bounds__(256) void flash_attn_kernel(const bf16_t* __restrict__ Q,
                                                         const bf16_t* __restrict__ K,
                                                         const bf16_t* __restrict__ V,
                                                         float* __restrict__ O) {
  __shared__ __align__(16) bf16_t Ks[32 * KSTRK];      // K rows, d-contiguous
  __shared__ __align__(16) bf16_t Vt[HD_ * VSTR];      // V transposed: [d][k]
  __shared__ __align__(16) bf16_t Pl[8 * 16 * PSTR];   // per-wave P re-layout

  const int tid  = threadIdx.x;
  const int lane = tid & 31;
  const int wave = tid >> 5;
  const int b  = blockIdx.z;
  const int h  = blockIdx.y;
  const int qb = blockIdx.x;
  const int q0w = qb * 128 + wave * 16;
  const int rstride = H_ * HD_;                // 2048

  const float SCALE = 0.08838834764831845f;    // 1/sqrt(128)

  const bf16_t* Qbase = Q + ((size_t)b * S_ + q0w) * rstride + h * HD_;
  v16bf qf[4];
#pragma unroll
  for (int c = 0; c < 4; ++c) qf[c] = frag16x32(Qbase + c * 32, rstride, lane);

  v8f o[8] = {};
  float rm[8], rl[8];
#pragma unroll
  for (int e = 0; e < 8; ++e) { rm[e] = NEG_BIG; rl[e] = 0.0f; }

  const int kend = qb * 128 + 128;
  const int kr = tid >> 3;                      // 0..31
  const int kd = (tid & 7) * 16;                // 0..112

  for (int kb = 0; kb < kend; kb += 32) {
    __syncthreads();
    // ---- K tile (32 x 128) via TDM ----
    if (wave == 0)
      tdm_load_2d(lds_off(&Ks[0]),
                  K + ((size_t)b * S_ + kb) * rstride + h * HD_,
                  128, 32, (unsigned)rstride, (unsigned)(B_ * S_),
                  (unsigned)rstride, 5, 3);
    // ---- V tile staged transposed (manual: TDM cannot transpose) ----
    const bf16_t* Vb = V + ((size_t)b * S_ + kb + kr) * rstride + h * HD_ + kd;
    v8bf v0 = *(const v8bf*)(Vb);
    v8bf v1 = *(const v8bf*)(Vb + 8);
#pragma unroll
    for (int i = 0; i < 8; ++i) {
      Vt[(kd + i) * VSTR + kr]     = v0[i];
      Vt[(kd + 8 + i) * VSTR + kr] = v1[i];
    }
    if (wave == 0) __builtin_amdgcn_s_wait_tensorcnt(0);
    __syncthreads();

    if (kb <= q0w + 15) {
      // ---- scores ----
      v8f s0 = {}, s1 = {};
#pragma unroll
      for (int c = 0; c < 4; ++c) {
        v16bf kf0 = frag16x32(&Ks[0  * KSTRK + c * 32], KSTRK, lane);
        v16bf kf1 = frag16x32(&Ks[16 * KSTRK + c * 32], KSTRK, lane);
        s0 = wmma_bf16(qf[c], kf0, s0);
        s1 = wmma_bf16(qf[c], kf1, s1);
      }

      // ---- online softmax ----
      const int half = lane >> 4, col = lane & 15;
#pragma unroll
      for (int e = 0; e < 8; ++e) {
        const int qr = q0w + 8 * half + e;
        float a0 = s0[e] * SCALE; if (kb + col      > qr) a0 = NEG_BIG;
        float a1 = s1[e] * SCALE; if (kb + 16 + col > qr) a1 = NEG_BIG;
        float mx = fmaxf(a0, a1);
#pragma unroll
        for (int d = 8; d >= 1; d >>= 1) mx = fmaxf(mx, __shfl_xor(mx, d, 32));
        const float mnew = fmaxf(rm[e], mx);
        const float p0 = __expf(a0 - mnew);
        const float p1 = __expf(a1 - mnew);
        float ps = p0 + p1;
#pragma unroll
        for (int d = 8; d >= 1; d >>= 1) ps += __shfl_xor(ps, d, 32);
        const float corr = __expf(rm[e] - mnew);
        rl[e] = rl[e] * corr + ps;
        rm[e] = mnew;
#pragma unroll
        for (int t = 0; t < 8; ++t) o[t][e] *= corr;
        bf16_t* prow = &Pl[(wave * 16 + 8 * half + e) * PSTR];
        prow[col]      = (bf16_t)p0;
        prow[col + 16] = (bf16_t)p1;
      }

      // ---- O += P @ V ----
      v16bf pf = frag16x32(&Pl[wave * 16 * PSTR], PSTR, lane);
#pragma unroll
      for (int t = 0; t < 8; ++t) {
        v16bf vf = frag16x32(&Vt[(t * 16) * VSTR], VSTR, lane);
        o[t] = wmma_bf16(pf, vf, o[t]);
      }
    }
  }

  // ---- normalize and store ----
  const int half = lane >> 4, col = lane & 15;
  float* ob = O + ((size_t)b * S_ + q0w + 8 * half) * rstride + h * HD_ + col;
#pragma unroll
  for (int e = 0; e < 8; ++e) {
    const float inv = 1.0f / rl[e];
#pragma unroll
    for (int t = 0; t < 8; ++t) ob[e * rstride + t * 16] = o[t][e] * inv;
  }
}

// ---------------------------------------------------------------- launch

extern "C" void kernel_launch(void* const* d_in, const int* in_sizes, int n_in,
                              void* d_out, int out_size, void* d_ws, size_t ws_size,
                              hipStream_t stream) {
  (void)in_sizes; (void)n_in; (void)out_size; (void)ws_size;

  const float* x    = (const float*)d_in[0];
  const float* fcos = (const float*)d_in[1];
  const float* fsin = (const float*)d_in[2];
  const float* wq   = (const float*)d_in[3];
  const float* wk   = (const float*)d_in[4];
  const float* wv   = (const float*)d_in[5];
  const float* wo   = (const float*)d_in[6];
  float* out = (float*)d_out;

  // Workspace layout (128 MB total)
  char* w = (char*)d_ws;
  size_t off = 0;
  bf16_t* xb  = (bf16_t*)(w + off); off += (size_t)M_ * D_ * 2;
  bf16_t* wqb = (bf16_t*)(w + off); off += (size_t)D_ * D_ * 2;
  bf16_t* wkb = (bf16_t*)(w + off); off += (size_t)D_ * D_ * 2;
  bf16_t* wvb = (bf16_t*)(w + off); off += (size_t)D_ * D_ * 2;
  bf16_t* wob = (bf16_t*)(w + off); off += (size_t)D_ * D_ * 2;
  float*  tmpF = (float*)(w + off); off += (size_t)M_ * D_ * 4;
  bf16_t* qb  = (bf16_t*)(w + off); off += (size_t)M_ * D_ * 2;
  bf16_t* kb  = (bf16_t*)(w + off); off += (size_t)M_ * D_ * 2;
  bf16_t* vb  = (bf16_t*)(w + off); off += (size_t)M_ * D_ * 2;

  const int nx = M_ * D_;
  const int nw = D_ * D_;

  f32_to_bf16_kernel<<<nx / 1024, 256, 0, stream>>>(x,  xb,  nx);
  f32_to_bf16_kernel<<<nw / 1024, 256, 0, stream>>>(wq, wqb, nw);
  f32_to_bf16_kernel<<<nw / 1024, 256, 0, stream>>>(wk, wkb, nw);
  f32_to_bf16_kernel<<<nw / 1024, 256, 0, stream>>>(wv, wvb, nw);
  f32_to_bf16_kernel<<<nw / 1024, 256, 0, stream>>>(wo, wob, nw);

  dim3 ggrid(D_ / 256, M_ / 128);   // (8, 32)
  const int npair = M_ * D_ / 2;

  gemm_bf16_kernel<<<ggrid, 256, 0, stream>>>(xb, wqb, tmpF);
  rotary_bf16_kernel<<<npair / 256, 256, 0, stream>>>(tmpF, fcos, fsin, qb);
  gemm_bf16_kernel<<<ggrid, 256, 0, stream>>>(xb, wkb, tmpF);
  rotary_bf16_kernel<<<npair / 256, 256, 0, stream>>>(tmpF, fcos, fsin, kb);
  gemm_bf16_kernel<<<ggrid, 256, 0, stream>>>(xb, wvb, tmpF);
  f32_to_bf16_kernel<<<nx / 1024, 256, 0, stream>>>(tmpF, vb, nx);

  dim3 fgrid(S_ / 128, H_, B_);     // (16, 16, 2)
  flash_attn_kernel<<<fgrid, 256, 0, stream>>>(qb, kb, vb, tmpF);

  f32_to_bf16_kernel<<<nx / 1024, 256, 0, stream>>>(tmpF, qb, nx);
  gemm_bf16_kernel<<<ggrid, 256, 0, stream>>>(qb, wob, out);
}
